// VAE_Model_84679575208310
// MI455X (gfx1250) — matrix operand order repaired and checked by reading, not compiled, not run on hardware
//
#include <hip/hip_runtime.h>
#include <stdint.h>
#include <math.h>

// ---------------- model dims ----------------
#define BATCH 1024
#define VOCABV 35
#define SEQV 110
#define NCKV 256
#define SCKV 9
#define SEFV 512
#define SLV 196
#define SDGV 512
#define SPV 256

typedef __attribute__((ext_vector_type(16))) __bf16 bf16x16;
typedef __attribute__((ext_vector_type(8)))  float  f32x8;

__device__ __forceinline__ uint16_t f2bf(float f) {
  union { float f; uint32_t u; } v; v.f = f;
  uint32_t u = v.u;
  return (uint16_t)((u + 0x7FFFu + ((u >> 16) & 1u)) >> 16); // RNE
}
__device__ __forceinline__ float bf2f(uint16_t h) {
  union { uint32_t u; float f; } v; v.u = ((uint32_t)h) << 16;
  return v.f;
}
__device__ __forceinline__ float sigmoidf_(float x) { return 1.0f / (1.0f + expf(-x)); }

// ---------------- generic bf16 WMMA GEMM ----------------
// C[M,N] = act( gather(A)[M,K] @ W[N,K]^T + bias[N] )
// MODE (compile-time):
//  0: row-major A[m*K+k] (f32)
//  1: conv, input layout [B,C,L]   (k = s*Cin + c)
//  2: conv, input layout [B,L,C]   (k = s*Cin + c)
//  3: encoder flatten of [B,L,C] with k = c*Lout + t   (channels-first flatten)
//  4: concat along K: k<K1 -> A[m*K1+k], else A2[m*(K-K1)+(k-K1)] (both f32)
// ACT: 0 none, 1 leaky(0.2), 2 softplus(x)+1e-6

#define BM 128
#define BN 128
#define BK 32
#define LDSS 48   // ushort stride (96 bytes: keeps 16B alignment, staggers banks)

struct GemmP {
  const void* A; const void* A2; const uint16_t* W; const float* bias;
  void* C;
  int M, N, K, K1, Cin, Lin, Lout;
};

template <int MODE, int ABF>
__device__ __forceinline__ float fetchA(const GemmP& p, int m, int k) {
  if (m >= p.M || k >= p.K) return 0.0f;
  long idx;
  if constexpr (MODE == 0) {
    idx = (long)m * p.K + k;
  } else if constexpr (MODE == 1) {
    int b = m / p.Lout, t = m - b * p.Lout;
    int s = k / p.Cin,  c = k - s * p.Cin;
    idx = ((long)b * p.Cin + c) * p.Lin + (t + s);
  } else if constexpr (MODE == 2) {
    int b = m / p.Lout, t = m - b * p.Lout;
    int s = k / p.Cin,  c = k - s * p.Cin;
    idx = ((long)b * p.Lin + (t + s)) * p.Cin + c;
  } else if constexpr (MODE == 3) {
    int c = k / p.Lout, t = k - c * p.Lout;
    idx = ((long)m * p.Lout + t) * p.Cin + c;
  } else { // MODE 4
    if (k < p.K1) idx = (long)m * p.K1 + k;
    else          return ((const float* __restrict__)p.A2)[(long)m * (p.K - p.K1) + (k - p.K1)];
  }
  if constexpr (ABF) return bf2f(((const uint16_t* __restrict__)p.A)[idx]);
  else               return ((const float* __restrict__)p.A)[idx];
}

template <int MODE, int ACT, int ABF, int CBF>
__global__ __launch_bounds__(256)
void gemm_wmma_bf16(GemmP p) {
  __shared__ uint16_t As[BM * LDSS];
  __shared__ uint16_t Bs[BN * LDSS];

  const int tid  = threadIdx.x;
  const int lane = tid & 31;
  const int wave = tid >> 5;
  const int wr   = wave >> 2;   // 0..1  -> 64-row strip
  const int wc   = wave & 3;    // 0..3  -> 32-col strip
  const int bm0  = blockIdx.y * BM;
  const int bn0  = blockIdx.x * BN;
  const int half = lane & 15;
  const int hi   = lane >> 4;

  f32x8 acc[4][2];
#pragma unroll
  for (int i = 0; i < 4; ++i)
#pragma unroll
    for (int j = 0; j < 2; ++j)
#pragma unroll
      for (int e = 0; e < 8; ++e) acc[i][j][e] = 0.0f;

  for (int k0 = 0; k0 < p.K; k0 += BK) {
    // ---- prefetch next weight k-tile (global_prefetch_b8 path) ----
    if (k0 + BK < p.K) {
      int pn = bn0 + (tid & 127);
      if (pn < p.N) __builtin_prefetch(&p.W[(long)pn * p.K + k0 + BK], 0, 1);
    }

    // ---- stage A tile ----
    if constexpr (MODE == 0 || MODE == 4) {
      // vectorized float4 path; mode 4 falls back per-chunk when crossing K1
#pragma unroll
      for (int it = 0; it < (BM * BK) / (4 * 256); ++it) {
        int i = tid + it * 256;
        int r = i >> 3, c4 = (i & 7) * 4;
        int m = bm0 + r, k = k0 + c4;
        float f0, f1, f2, f3;
        bool vec_ok = (m < p.M) && (k + 4 <= p.K) &&
                      (MODE == 0 || (k + 4 <= p.K1) || (k >= p.K1));
        if (vec_ok) {
          const float* __restrict__ src;
          if (MODE == 4 && k >= p.K1)
            src = (const float*)p.A2 + (long)m * (p.K - p.K1) + (k - p.K1);
          else
            src = (const float*)p.A + (long)m * (MODE == 4 ? p.K1 : p.K) + k;
          float4 f = *(const float4*)src;
          f0 = f.x; f1 = f.y; f2 = f.z; f3 = f.w;
        } else {
          f0 = fetchA<MODE, ABF>(p, m, k);
          f1 = fetchA<MODE, ABF>(p, m, k + 1);
          f2 = fetchA<MODE, ABF>(p, m, k + 2);
          f3 = fetchA<MODE, ABF>(p, m, k + 3);
        }
        union { uint16_t s[4]; uint2 v; } o;
        o.s[0] = f2bf(f0); o.s[1] = f2bf(f1); o.s[2] = f2bf(f2); o.s[3] = f2bf(f3);
        *(uint2*)&As[r * LDSS + c4] = o.v;
      }
    } else {
      // gather path (implicit im2col / flatten), scalar with f32->bf16 convert
#pragma unroll 4
      for (int i = tid; i < BM * BK; i += 256) {
        int r = i >> 5, c = i & 31;
        As[r * LDSS + c] = f2bf(fetchA<MODE, ABF>(p, bm0 + r, k0 + c));
      }
    }

    // ---- stage B tile: bf16 weights row-major [N,K], 16B vector loads ----
#pragma unroll
    for (int it = 0; it < (BN * BK) / (8 * 256); ++it) {
      int i = tid + it * 256;
      int r = i >> 2, c8 = (i & 3) * 8;
      int n = bn0 + r, k = k0 + c8;
      uint4 val;
      if (n < p.N && k + 8 <= p.K) {
        val = *(const uint4*)&p.W[(long)n * p.K + k];
      } else {
        union { uint16_t s[8]; uint4 v; } o;
#pragma unroll
        for (int e = 0; e < 8; ++e)
          o.s[e] = (n < p.N && k + e < p.K) ? p.W[(long)n * p.K + k + e] : (uint16_t)0;
        val = o.v;
      }
      *(uint4*)&Bs[r * LDSS + c8] = val;
    }
    __syncthreads();

    // ---- B fragments: lane half selects K 0..15 / 16..31, contiguous 32B ----
    bf16x16 bfr[2];
#pragma unroll
    for (int j = 0; j < 2; ++j) {
      const uint16_t* base = &Bs[(wc * 32 + j * 16 + half) * LDSS + hi * 16];
      union { bf16x16 v; uint4 u[2]; } t;
      t.u[0] = *(const uint4*)(base);
      t.u[1] = *(const uint4*)(base + 8);
      bfr[j] = t.v;
    }
    // ---- A fragments: lane half selects K 0..15 / 16..31, contiguous 32B ----
#pragma unroll
    for (int i = 0; i < 4; ++i) {
      const uint16_t* base = &As[(wr * 64 + i * 16 + half) * LDSS];
      union { bf16x16 v; uint4 u[2]; } t;
      t.u[0] = *(const uint4*)(base + hi * 16);
      t.u[1] = *(const uint4*)(base + hi * 16 + 8);
      bf16x16 afr = t.v;
#pragma unroll
      for (int j = 0; j < 2; ++j) {
        acc[i][j] = __builtin_amdgcn_wmma_f32_16x16x32_bf16(
            false, afr, false, bfr[j], (short)0, acc[i][j], false, false);
      }
    }
    __syncthreads();
  }

  // ---- epilogue: bias + activation + store (f32 or bf16) ----
#pragma unroll
  for (int i = 0; i < 4; ++i) {
#pragma unroll
    for (int j = 0; j < 2; ++j) {
      int n = bn0 + wc * 32 + j * 16 + half;
      if (n >= p.N) continue;
      float bv = p.bias ? p.bias[n] : 0.0f;
#pragma unroll
      for (int v = 0; v < 8; ++v) {
        int m = bm0 + wr * 64 + i * 16 + hi * 8 + v;
        if (m >= p.M) continue;
        float x = acc[i][j][v] + bv;
        if constexpr (ACT == 1) x = (x > 0.0f) ? x : 0.2f * x;
        else if constexpr (ACT == 2) { x = (x > 20.0f) ? x : log1pf(expf(x)); x += 1e-6f; }
        long o = (long)m * p.N + n;
        if constexpr (CBF) ((uint16_t*)p.C)[o] = f2bf(x);
        else               ((float*)p.C)[o]    = x;
      }
    }
  }
}

// ---------------- elementwise kernels ----------------
__global__ void k_zero_f32(float* p, long n) {
  long i = (long)blockIdx.x * blockDim.x + threadIdx.x;
  if (i < n) p[i] = 0.0f;
}
__global__ void k_lin_w_bf16(const float* w, uint16_t* o, long n) {
  long i = (long)blockIdx.x * blockDim.x + threadIdx.x;
  if (i < n) o[i] = f2bf(w[i]);
}
// permute conv weight [Cout,Cin,S] -> bf16 [Cout, S*Cin] with k = s*Cin + c
__global__ void k_conv_w_bf16(const float* w, uint16_t* o, int Cout, int Cin, int S) {
  long i = (long)blockIdx.x * blockDim.x + threadIdx.x;
  long tot = (long)Cout * Cin * S;
  if (i >= tot) return;
  int n = (int)(i / ((long)Cin * S));
  int rem = (int)(i - (long)n * Cin * S);
  int c = rem / S, s = rem - c * S;
  o[(long)n * Cin * S + (long)s * Cin + c] = f2bf(w[i]);
}
// d = std*eps + mu
__global__ void k_reparam(const float* mu, const float* sd, const float* eps, float* d, long n) {
  long i = (long)blockIdx.x * blockDim.x + threadIdx.x;
  if (i < n) d[i] = sd[i] * eps[i] + mu[i];
}
// GRU gate update, gate order [r,z,n]; h updated in place
__global__ void k_gru_gate(const float* __restrict__ gi, const float* __restrict__ gh,
                           float* h, int H) {
  long i = (long)blockIdx.x * blockDim.x + threadIdx.x;
  if (i >= (long)BATCH * H) return;
  int b = (int)(i / H), j = (int)(i - (long)b * H);
  const float* gib = gi + (long)b * 3 * H;
  const float* ghb = gh + (long)b * 3 * H;
  float r = sigmoidf_(gib[j]         + ghb[j]);
  float z = sigmoidf_(gib[H + j]     + ghb[H + j]);
  float n = tanhf    (gib[2 * H + j] + r * ghb[2 * H + j]);
  h[i] = (1.0f - z) * n + z * h[i];
}
// term GRU cell with h==0: gh = bhh; h' = (1-z)*n ; optionally scatter into term_out[b,j,t]
__global__ void k_term_gate(const float* __restrict__ gi, const float* __restrict__ bhh,
                            float* hout, float* term_out, int t) {
  long i = (long)blockIdx.x * blockDim.x + threadIdx.x;
  if (i >= (long)BATCH * VOCABV) return;
  int b = (int)(i / VOCABV), j = (int)(i - (long)b * VOCABV);
  const float* g = gi + (long)b * 3 * VOCABV;
  float r = sigmoidf_(g[j]              + bhh[j]);
  float z = sigmoidf_(g[VOCABV + j]     + bhh[VOCABV + j]);
  float n = tanhf    (g[2 * VOCABV + j] + r * bhh[2 * VOCABV + j]);
  float h = (1.0f - z) * n;
  hout[i] = h;
  if (term_out) term_out[((long)b * VOCABV + j) * SEQV + t] = h;
}

// ---------------- host-side helpers ----------------
template <int MODE, int ACT, int ABF, int CBF>
static inline void gemm(hipStream_t s, const void* A, const void* A2,
                        const uint16_t* W, const float* bias, void* C,
                        int M, int N, int K,
                        int K1 = 0, int Cin = 0, int Lin = 0, int Lout = 0) {
  GemmP p{A, A2, W, bias, C, M, N, K, K1, Cin, Lin, Lout};
  dim3 g((unsigned)((N + BN - 1) / BN), (unsigned)((M + BM - 1) / BM));
  gemm_wmma_bf16<MODE, ACT, ABF, CBF><<<g, dim3(256), 0, s>>>(p);
}
static inline void conv_w(hipStream_t s, const float* w, uint16_t* o, int Cout, int Cin) {
  long n = (long)Cout * Cin * SCKV;
  k_conv_w_bf16<<<dim3((unsigned)((n + 255) / 256)), dim3(256), 0, s>>>(w, o, Cout, Cin, SCKV);
}
static inline void lin_w(hipStream_t s, const float* w, uint16_t* o, long n) {
  k_lin_w_bf16<<<dim3((unsigned)((n + 255) / 256)), dim3(256), 0, s>>>(w, o, n);
}
static inline void zero(hipStream_t s, float* p, long n) {
  k_zero_f32<<<dim3((unsigned)((n + 255) / 256)), dim3(256), 0, s>>>(p, n);
}

extern "C" void kernel_launch(void* const* d_in, const int* in_sizes, int n_in,
                              void* d_out, int out_size, void* d_ws, size_t ws_size,
                              hipStream_t stream) {
  (void)in_sizes; (void)n_in; (void)out_size; (void)ws_size;
  // -------- inputs (leaf order of setup_inputs dict) --------
  const float* model_input = (const float*)d_in[0];
  const float* epsilon     = (const float*)d_in[1];
  const float* Wc[3] = {(const float*)d_in[2], (const float*)d_in[4], (const float*)d_in[6]};
  const float* bc[3] = {(const float*)d_in[3], (const float*)d_in[5], (const float*)d_in[7]};
  const float* We[3] = {(const float*)d_in[8], (const float*)d_in[10], (const float*)d_in[12]};
  const float* be[3] = {(const float*)d_in[9], (const float*)d_in[11], (const float*)d_in[13]};
  const float* Wmu = (const float*)d_in[14]; const float* bmu = (const float*)d_in[15];
  const float* Wsd = (const float*)d_in[16]; const float* bsd = (const float*)d_in[17];
  const float* Wd[3] = {(const float*)d_in[18], (const float*)d_in[20], (const float*)d_in[22]};
  const float* bd[3] = {(const float*)d_in[19], (const float*)d_in[21], (const float*)d_in[23]};
  const float *Wih[3], *Whh[3], *bih[3], *bhh[3];
  for (int l = 0; l < 3; ++l) {
    Wih[l] = (const float*)d_in[24 + 4 * l]; Whh[l] = (const float*)d_in[25 + 4 * l];
    bih[l] = (const float*)d_in[26 + 4 * l]; bhh[l] = (const float*)d_in[27 + 4 * l];
  }
  const float *tWih[3], *tbih[3], *tbhh[3];
  for (int l = 0; l < 3; ++l) {
    tWih[l] = (const float*)d_in[36 + 4 * l];
    tbih[l] = (const float*)d_in[38 + 4 * l]; tbhh[l] = (const float*)d_in[39 + 4 * l];
  }
  const float *Wp[4], *bp[4];
  for (int l = 0; l < 4; ++l) { Wp[l] = (const float*)d_in[48 + 2 * l]; bp[l] = (const float*)d_in[49 + 2 * l]; }

  // -------- conv geometry --------
  const int L1 = SEQV - (SCKV - 1);        // 102
  const int L2 = L1 - (SCKV - 1);          // 94
  const int L3 = L2 - (SCKV - 1);          // 86
  const int K_c1 = VOCABV * SCKV;          // 315
  const int K_c  = NCKV * SCKV;            // 2304
  const int FLATK = NCKV * L3;             // 22016
  const int GH = 3 * SDGV;                 // 1536
  const int TK1 = SDGV + VOCABV;           // 547
  const int TN  = 3 * VOCABV;              // 105

  // -------- workspace bump allocator --------
  size_t off = 0;
  auto alloc = [&](size_t bytes) -> void* {
    void* p = (char*)d_ws + off;
    off += (bytes + 255) & ~(size_t)255;
    return p;
  };
  // bf16 weights
  uint16_t* wc1 = (uint16_t*)alloc((size_t)NCKV * K_c1 * 2);
  uint16_t* wc2 = (uint16_t*)alloc((size_t)NCKV * K_c * 2);
  uint16_t* wc3 = (uint16_t*)alloc((size_t)NCKV * K_c * 2);
  uint16_t* we1 = (uint16_t*)alloc((size_t)SEFV * FLATK * 2);
  uint16_t* we2 = (uint16_t*)alloc((size_t)SEFV * SEFV * 2);
  uint16_t* we3 = (uint16_t*)alloc((size_t)SEFV * SEFV * 2);
  uint16_t* wmu = (uint16_t*)alloc((size_t)SLV * SEFV * 2);
  uint16_t* wsd = (uint16_t*)alloc((size_t)SLV * SEFV * 2);
  uint16_t* wd1 = (uint16_t*)alloc((size_t)SEFV * SLV * 2);
  uint16_t* wd2 = (uint16_t*)alloc((size_t)SEFV * SEFV * 2);
  uint16_t* wd3 = (uint16_t*)alloc((size_t)SEFV * SEFV * 2);
  uint16_t *wih[3], *whh[3];
  for (int l = 0; l < 3; ++l) {
    wih[l] = (uint16_t*)alloc((size_t)GH * SDGV * 2);
    whh[l] = (uint16_t*)alloc((size_t)GH * SDGV * 2);
  }
  uint16_t* twih0 = (uint16_t*)alloc((size_t)TN * TK1 * 2);
  uint16_t* twih1 = (uint16_t*)alloc((size_t)TN * VOCABV * 2);
  uint16_t* twih2 = (uint16_t*)alloc((size_t)TN * VOCABV * 2);
  uint16_t* wp1 = (uint16_t*)alloc((size_t)SPV * SLV * 2);
  uint16_t* wp2 = (uint16_t*)alloc((size_t)SPV * SPV * 2);
  uint16_t* wp3 = (uint16_t*)alloc((size_t)SPV * SPV * 2);
  uint16_t* wp4 = (uint16_t*)alloc((size_t)3 * SPV * 2);
  // activations
  uint16_t* y1 = (uint16_t*)alloc((size_t)BATCH * L1 * NCKV * 2); // conv1 out [B,L1,C] bf16
  uint16_t* y2 = (uint16_t*)alloc((size_t)BATCH * L2 * NCKV * 2); // conv2 out
  uint16_t* y3 = y1;                                              // conv3 out reuses y1 (fits)
  float* ha   = (float*)alloc((size_t)BATCH * SEFV * 4);
  float* hb   = (float*)alloc((size_t)BATCH * SEFV * 4);
  float* dlat = (float*)alloc((size_t)BATCH * SLV * 4);
  float* xdec = (float*)alloc((size_t)BATCH * SEFV * 4);
  float* gi1  = (float*)alloc((size_t)BATCH * GH * 4);
  float* gib  = (float*)alloc((size_t)BATCH * GH * 4);
  float* ghb  = (float*)alloc((size_t)BATCH * GH * 4);
  float* h1   = (float*)alloc((size_t)BATCH * SDGV * 4);
  float* h2   = (float*)alloc((size_t)BATCH * SDGV * 4);
  float* h3   = (float*)alloc((size_t)BATCH * SDGV * 4);
  float* tgi  = (float*)alloc((size_t)BATCH * TN * 4);
  float* th   = (float*)alloc((size_t)BATCH * VOCABV * 4);
  float* th2  = (float*)alloc((size_t)BATCH * VOCABV * 4);
  float* tprv = (float*)alloc((size_t)BATCH * VOCABV * 4);

  // -------- output layout (mu, std, term_out, p) --------
  float* out_mu   = (float*)d_out;
  float* out_std  = out_mu + (size_t)BATCH * SLV;
  float* out_term = out_std + (size_t)BATCH * SLV;
  float* out_p    = out_term + (size_t)BATCH * VOCABV * SEQV;

  // -------- 1) weight conversion to bf16 (once per launch) --------
  conv_w(stream, Wc[0], wc1, NCKV, VOCABV);
  conv_w(stream, Wc[1], wc2, NCKV, NCKV);
  conv_w(stream, Wc[2], wc3, NCKV, NCKV);
  lin_w(stream, We[0], we1, (long)SEFV * FLATK);
  lin_w(stream, We[1], we2, (long)SEFV * SEFV);
  lin_w(stream, We[2], we3, (long)SEFV * SEFV);
  lin_w(stream, Wmu, wmu, (long)SLV * SEFV);
  lin_w(stream, Wsd, wsd, (long)SLV * SEFV);
  lin_w(stream, Wd[0], wd1, (long)SEFV * SLV);
  lin_w(stream, Wd[1], wd2, (long)SEFV * SEFV);
  lin_w(stream, Wd[2], wd3, (long)SEFV * SEFV);
  for (int l = 0; l < 3; ++l) {
    lin_w(stream, Wih[l], wih[l], (long)GH * SDGV);
    lin_w(stream, Whh[l], whh[l], (long)GH * SDGV);
  }
  lin_w(stream, tWih[0], twih0, (long)TN * TK1);
  lin_w(stream, tWih[1], twih1, (long)TN * VOCABV);
  lin_w(stream, tWih[2], twih2, (long)TN * VOCABV);
  lin_w(stream, Wp[0], wp1, (long)SPV * SLV);
  lin_w(stream, Wp[1], wp2, (long)SPV * SPV);
  lin_w(stream, Wp[2], wp3, (long)SPV * SPV);
  lin_w(stream, Wp[3], wp4, (long)3 * SPV);

  // -------- 2) conv stack as implicit-im2col WMMA GEMMs (LeakyReLU fused) --------
  gemm<1, 1, 0, 1>(stream, model_input, nullptr, wc1, bc[0], y1,
                   BATCH * L1, NCKV, K_c1, 0, VOCABV, SEQV, L1);
  gemm<2, 1, 1, 1>(stream, y1, nullptr, wc2, bc[1], y2,
                   BATCH * L2, NCKV, K_c, 0, NCKV, L1, L2);
  gemm<2, 1, 1, 1>(stream, y2, nullptr, wc3, bc[2], y3,
                   BATCH * L3, NCKV, K_c, 0, NCKV, L2, L3);

  // -------- 3) encoder FNN (mode 3 gathers channels-first flatten) --------
  gemm<3, 1, 1, 0>(stream, y3, nullptr, we1, be[0], ha, BATCH, SEFV, FLATK,
                   0, NCKV, 0, L3);
  gemm<0, 1, 0, 0>(stream, ha, nullptr, we2, be[1], hb, BATCH, SEFV, SEFV);
  gemm<0, 1, 0, 0>(stream, hb, nullptr, we3, be[2], ha, BATCH, SEFV, SEFV);

  // -------- 4) mu / std (softplus+1e-6 fused) straight into d_out --------
  gemm<0, 0, 0, 0>(stream, ha, nullptr, wmu, bmu, out_mu,  BATCH, SLV, SEFV);
  gemm<0, 2, 0, 0>(stream, ha, nullptr, wsd, bsd, out_std, BATCH, SLV, SEFV);
  {
    long n = (long)BATCH * SLV;
    k_reparam<<<dim3((unsigned)((n + 255) / 256)), dim3(256), 0, stream>>>(
        out_mu, out_std, epsilon, dlat, n);
  }

  // -------- 5) decoder FNN --------
  gemm<0, 1, 0, 0>(stream, dlat, nullptr, wd1, bd[0], hb, BATCH, SEFV, SLV);
  gemm<0, 1, 0, 0>(stream, hb, nullptr, wd2, bd[1], ha, BATCH, SEFV, SEFV);
  gemm<0, 1, 0, 0>(stream, ha, nullptr, wd3, bd[2], xdec, BATCH, SEFV, SEFV);

  // -------- 6) fused temporal loop: 3-layer GRU + autoregressive term chain --------
  zero(stream, h1, (long)BATCH * SDGV);
  zero(stream, h2, (long)BATCH * SDGV);
  zero(stream, h3, (long)BATCH * SDGV);
  zero(stream, tprv, (long)BATCH * VOCABV);

  // layer-1 input is the broadcast decoder vector: gi is loop-invariant
  gemm<0, 0, 0, 0>(stream, xdec, nullptr, wih[0], bih[0], gi1, BATCH, GH, SDGV);

  const unsigned gruBlocks = (unsigned)(((long)BATCH * SDGV + 255) / 256);
  const unsigned trmBlocks = (unsigned)(((long)BATCH * VOCABV + 255) / 256);

  for (int t = 0; t < SEQV; ++t) {
    // layer 1
    gemm<0, 0, 0, 0>(stream, h1, nullptr, whh[0], bhh[0], ghb, BATCH, GH, SDGV);
    k_gru_gate<<<dim3(gruBlocks), dim3(256), 0, stream>>>(gi1, ghb, h1, SDGV);
    // layer 2
    gemm<0, 0, 0, 0>(stream, h1, nullptr, wih[1], bih[1], gib, BATCH, GH, SDGV);
    gemm<0, 0, 0, 0>(stream, h2, nullptr, whh[1], bhh[1], ghb, BATCH, GH, SDGV);
    k_gru_gate<<<dim3(gruBlocks), dim3(256), 0, stream>>>(gib, ghb, h2, SDGV);
    // layer 3
    gemm<0, 0, 0, 0>(stream, h2, nullptr, wih[2], bih[2], gib, BATCH, GH, SDGV);
    gemm<0, 0, 0, 0>(stream, h3, nullptr, whh[2], bhh[2], ghb, BATCH, GH, SDGV);
    k_gru_gate<<<dim3(gruBlocks), dim3(256), 0, stream>>>(gib, ghb, h3, SDGV);

    // term chain (fresh zero hidden each cell -> gh == bhh, Whh GEMMs eliminated)
    gemm<4, 0, 0, 0>(stream, h3, tprv, twih0, tbih[0], tgi, BATCH, TN, TK1, /*K1*/SDGV);
    k_term_gate<<<dim3(trmBlocks), dim3(256), 0, stream>>>(tgi, tbhh[0], th, (float*)nullptr, t);
    gemm<0, 0, 0, 0>(stream, th, nullptr, twih1, tbih[1], tgi, BATCH, TN, VOCABV);
    k_term_gate<<<dim3(trmBlocks), dim3(256), 0, stream>>>(tgi, tbhh[1], th2, (float*)nullptr, t);
    gemm<0, 0, 0, 0>(stream, th2, nullptr, twih2, tbih[2], tgi, BATCH, TN, VOCABV);
    k_term_gate<<<dim3(trmBlocks), dim3(256), 0, stream>>>(tgi, tbhh[2], tprv, out_term, t);
  }

  // -------- 7) predictor head from mu --------
  gemm<0, 1, 0, 0>(stream, out_mu, nullptr, wp1, bp[0], ha, BATCH, SPV, SLV);
  gemm<0, 1, 0, 0>(stream, ha, nullptr, wp2, bp[1], hb, BATCH, SPV, SPV);
  gemm<0, 1, 0, 0>(stream, hb, nullptr, wp3, bp[2], ha, BATCH, SPV, SPV);
  gemm<0, 0, 0, 0>(stream, ha, nullptr, wp4, bp[3], out_p, BATCH, 3, SPV);
}
